// EnhancedSNNCifar_7610682048754
// MI455X (gfx1250) — compile-verified, matrix operand order checked by reading, not run on hardware
//
#include <hip/hip_runtime.h>
#include <hip/hip_bf16.h>

// ============================================================================
// Spiking CNN (T=8, N=128) forward for MI455X / gfx1250.
//  - convs 2..6 + FC1: implicit-GEMM with v_wmma_f32_16x16x32_f16 (wave32),
//    A = binary spikes (exact in f16), B = f16-packed weights, f32 accum.
//    Block tile 128x64, wave tile 32x32 -> 4 wmma per K-step per wave.
//  - global->LDS staging uses CDNA5 async load-to-LDS (ASYNCcnt) when the
//    toolchain exposes the builtins, else load+ds_store fallback.
//  - training-mode BN: two-pass (f32 pre-act -> channel stats -> fused BN+LIF
//    scan over T emitting f16 spikes). Conv bias cancels under BN -> dropped.
//  - layer 1 (3->32, broadcast over T): computed once, direct conv.
//  - NHWC activation layout so im2col patches are contiguous 16B chunks.
// ============================================================================

typedef __attribute__((ext_vector_type(16))) _Float16 v16h;
typedef __attribute__((ext_vector_type(8)))  float    v8f;
typedef int v4i __attribute__((vector_size(16)));   // matches builtin param type

#define TSTEPS 8
#define NBATCH 128
#define TN     (TSTEPS * NBATCH)   // 1024

// ---------------------------------------------------------------------------
// async global->LDS (gfx1250): guarded, with synchronous fallback
// ---------------------------------------------------------------------------
#if defined(__has_builtin)
#  if __has_builtin(__builtin_amdgcn_global_load_async_to_lds_b128) && \
      __has_builtin(__builtin_amdgcn_s_wait_asynccnt)
#    define USE_ASYNC_LDS 1
#  endif
#endif
#ifndef USE_ASYNC_LDS
#  define USE_ASYNC_LDS 0
#endif

// copy one 16-byte chunk into LDS (each lane owns its chunk exclusively);
// ok==false -> zero-fill (im2col padding / channel tail)
__device__ __forceinline__ void copy16_to_lds(_Float16* dst, const _Float16* src,
                                              bool ok) {
#if USE_ASYNC_LDS
  if (ok) {
    __builtin_amdgcn_global_load_async_to_lds_b128(
        (__attribute__((address_space(1))) v4i*)(void*)src,
        (__attribute__((address_space(3))) v4i*)(void*)dst, 0, 0);
  } else {
    *(uint4*)dst = make_uint4(0u, 0u, 0u, 0u);
  }
#else
  uint4 v = make_uint4(0u, 0u, 0u, 0u);
  if (ok) v = *(const uint4*)src;
  *(uint4*)dst = v;
#endif
}

__device__ __forceinline__ void lds_copy_fence() {
#if USE_ASYNC_LDS
  __builtin_amdgcn_s_wait_asynccnt(0);   // s_wait_asynccnt 0
#endif
}

// ---------------------------------------------------------------------------
// tiny zero kernel (graph-capture-safe replacement for memset)
// ---------------------------------------------------------------------------
__global__ void zero_f32(float* __restrict__ p, int n) {
  int i = blockIdx.x * 256 + threadIdx.x;
  if (i < n) p[i] = 0.f;
}

// ---------------------------------------------------------------------------
// weight packing: conv [O,I,3,3] f32 -> [O][K=9*I] f16, k = (kh*3+kw)*I + ci
// ---------------------------------------------------------------------------
__global__ void pack_conv_w(const float* __restrict__ Wsrc,
                            _Float16* __restrict__ WT, int O, int I) {
  int K = 9 * I;
  int idx = blockIdx.x * 256 + threadIdx.x;
  if (idx >= O * K) return;
  int o = idx / K, k = idx - o * K;
  int p = k / I, ci = k - p * I;
  int kh = p / 3, kw = p - kh * 3;
  WT[idx] = (_Float16)Wsrc[((o * I + ci) * 3 + kh) * 3 + kw];
}

// fc1_w [128,2048] with reference k = c*16 + h*4 + w  ->  NHWC flatten order
__global__ void pack_fc1_w(const float* __restrict__ Wsrc,
                           _Float16* __restrict__ WT) {
  int idx = blockIdx.x * 256 + threadIdx.x;
  if (idx >= 128 * 2048) return;
  int o  = idx >> 11;
  int kp = idx & 2047;           // (h*4+w)*128 + c
  int c  = kp & 127;
  int hw = kp >> 7;              // h*4+w
  WT[idx] = (_Float16)Wsrc[o * 2048 + c * 16 + hw];
}

// ---------------------------------------------------------------------------
// layer 1: direct conv 3->32, pad 1, N=128 images (broadcast over T).
// X NCHW f32, Y NHWC f32 [128,32,32,32]
// ---------------------------------------------------------------------------
__global__ __launch_bounds__(256)
void conv1_direct(const float* __restrict__ X, const float* __restrict__ Wc,
                  float* __restrict__ Y) {
  __shared__ float w[32 * 27];
  for (int i = threadIdx.x; i < 32 * 27; i += 256) w[i] = Wc[i];
  __syncthreads();
  int idx = blockIdx.x * 256 + threadIdx.x;     // (n, oh, ow, co)
  int co = idx & 31; int r = idx >> 5;
  int ow = r & 31; r >>= 5;
  int oh = r & 31; int n = r >> 5;
  if (n >= NBATCH) return;
  float acc = 0.f;
  #pragma unroll
  for (int ci = 0; ci < 3; ci++)
    #pragma unroll
    for (int kh = 0; kh < 3; kh++) {
      int ih = oh + kh - 1;
      if (ih < 0 || ih > 31) continue;
      #pragma unroll
      for (int kw = 0; kw < 3; kw++) {
        int iw = ow + kw - 1;
        if (iw < 0 || iw > 31) continue;
        acc += X[((n * 3 + ci) * 32 + ih) * 32 + iw] * w[co * 27 + ci * 9 + kh * 3 + kw];
      }
    }
  Y[idx] = acc;   // NHWC
}

// ---------------------------------------------------------------------------
// implicit-GEMM 3x3 conv with WMMA.  S: spikes [TN,H,W,CIN] f16 (NHWC),
// WT: [COUT][K=9*CIN] f16, Y: [M=TN*H*W, COUT] f32.
// Block 256thr = 8 waves; tile M=128 x N=NB (32|64); K stepped by 32.
// Wave (wm,wn): rows wm*32..+32 (2 A frags), cols wn*(NB/2)..+(NB/2).
// ---------------------------------------------------------------------------
template <int CIN, int COUT>
__global__ __launch_bounds__(256)
void conv3x3_wmma(const _Float16* __restrict__ S,
                  const _Float16* __restrict__ WT,
                  float* __restrict__ Y, int M, int H, int W) {
  constexpr int K  = 9 * CIN;
  constexpr int NB = (COUT < 64) ? COUT : 64;   // 32 or 64
  constexpr int NT = NB / 32;                   // B frags per wave (1|2)
  constexpr int LDA = 40;                       // halfs/row in LDS (+8 pad)

  __shared__ _Float16 lA[128 * LDA];
  __shared__ _Float16 lB[64 * LDA];

  const int tid  = threadIdx.x;
  const int wave = tid >> 5, lane = tid & 31;
  const int wm = wave & 3;            // row group of 32
  const int wn = wave >> 2;           // col half
  const int mblk = blockIdx.x * 128;
  const int nblk = blockIdx.y * 64;

  v8f acc[2][NT] = {};

  // cooperative loaders: A = 512 x 16B chunks (2/thread), B = 256 (1/thread)
  const int lro = tid >> 2;           // 0..63
  const int lko = (tid & 3) * 8;      // 0,8,16,24 halfs
  int n_img[2], oh[2], ow[2];
  bool rowok[2];
  #pragma unroll
  for (int u = 0; u < 2; u++) {
    int gpix = mblk + lro + u * 64;
    rowok[u] = gpix < M;
    n_img[u] = 0; oh[u] = 0; ow[u] = 0;
    if (rowok[u]) {
      n_img[u] = gpix / (H * W);
      int r = gpix - n_img[u] * H * W;
      oh[u] = r / W; ow[u] = r - oh[u] * W;
    }
  }
  const int bcol = nblk + lro;

  for (int k0 = 0; k0 < K; k0 += 32) {
    int k  = k0 + lko;
    int p  = k / CIN, ci = k - p * CIN;   // 32|CIN -> chunk within one cell
    int kh = p / 3,   kw = p - kh * 3;

    __syncthreads();                      // prev compute done before overwrite
    #pragma unroll
    for (int u = 0; u < 2; u++) {
      int ih = oh[u] + kh - 1, iw = ow[u] + kw - 1;
      bool ok = rowok[u] && ih >= 0 && ih < H && iw >= 0 && iw < W;
      copy16_to_lds(lA + (lro + u * 64) * LDA + lko,
                    S + ((size_t)(n_img[u] * H + ih) * W + iw) * CIN + ci, ok);
    }
    copy16_to_lds(lB + lro * LDA + lko,
                  WT + (size_t)bcol * K + k0 + lko, bcol < COUT);
    if (bcol < COUT && k0 + 32 < K)
      __builtin_prefetch(WT + (size_t)bcol * K + k0 + 32 + lko, 0, 0);
    lds_copy_fence();
    __syncthreads();

    // fragments (ISA 16-bit 16x32 layout: lanes 0-15 hold K 0-7 & 16-23,
    // lanes 16-31 hold K 8-15 & 24-31)
    const int ls = lane >> 4, r = lane & 15;
    v16h af[2];
    #pragma unroll
    for (int f = 0; f < 2; f++) {
      const _Float16* pa = lA + (wm * 32 + f * 16 + r) * LDA + ls * 8;
      ((uint4*)&af[f])[0] = *(const uint4*)(pa);
      ((uint4*)&af[f])[1] = *(const uint4*)(pa + 16);
    }
    #pragma unroll
    for (int s = 0; s < NT; s++) {
      v16h bf;
      const _Float16* pb = lB + (wn * (NB / 2) + s * 16 + r) * LDA + ls * 8;
      ((uint4*)&bf)[0] = *(const uint4*)(pb);
      ((uint4*)&bf)[1] = *(const uint4*)(pb + 16);
      #pragma unroll
      for (int f = 0; f < 2; f++)
        acc[f][s] = __builtin_amdgcn_wmma_f32_16x16x32_f16(
            false, af[f], false, bf, (short)0, acc[f][s], false, false);
    }
  }

  // epilogue: C/D f32 layout: lane%16 = column, vgpr j -> row (lane/16)*8+j
  const int ls = lane >> 4, nl = lane & 15;
  #pragma unroll
  for (int f = 0; f < 2; f++)
    #pragma unroll
    for (int s = 0; s < NT; s++) {
      int o = nblk + wn * (NB / 2) + s * 16 + nl;
      if (o < COUT) {
        #pragma unroll
        for (int j = 0; j < 8; j++) {
          int row = mblk + wm * 32 + f * 16 + ls * 8 + j;
          if (row < M) Y[(size_t)row * COUT + o] = acc[f][s][j];
        }
      }
    }
}

// ---------------------------------------------------------------------------
// FC1 WMMA GEMM: F [1024, 2048] f16 (flattened pooled spikes, NHWC order),
// WT [128][2048] f16, bias f32 -> Y [1024,128] f32.  Tile 128x64.
// ---------------------------------------------------------------------------
__global__ __launch_bounds__(256)
void fc1_wmma(const _Float16* __restrict__ F, const _Float16* __restrict__ WT,
              const float* __restrict__ bias, float* __restrict__ Y) {
  constexpr int K = 2048, COUT = 128, LDA = 40;
  __shared__ _Float16 lA[128 * LDA];
  __shared__ _Float16 lB[64 * LDA];
  const int tid = threadIdx.x, wave = tid >> 5, lane = tid & 31;
  const int wm = wave & 3, wn = wave >> 2;
  const int mblk = blockIdx.x * 128, nblk = blockIdx.y * 64;
  v8f acc[2][2] = {};
  const int lro = tid >> 2, lko = (tid & 3) * 8;
  const int bcol = nblk + lro;          // < 128 by construction
  for (int k0 = 0; k0 < K; k0 += 32) {
    __syncthreads();
    #pragma unroll
    for (int u = 0; u < 2; u++) {
      int arow = mblk + lro + u * 64;   // < 1024 by construction
      copy16_to_lds(lA + (lro + u * 64) * LDA + lko,
                    F + (size_t)arow * K + k0 + lko, true);
      if (k0 + 32 < K)
        __builtin_prefetch(F + (size_t)arow * K + k0 + 32 + lko, 0, 0);
    }
    copy16_to_lds(lB + lro * LDA + lko, WT + (size_t)bcol * K + k0 + lko, true);
    if (k0 + 32 < K)
      __builtin_prefetch(WT + (size_t)bcol * K + k0 + 32 + lko, 0, 0);
    lds_copy_fence();
    __syncthreads();

    const int ls = lane >> 4, r = lane & 15;
    v16h af[2];
    #pragma unroll
    for (int f = 0; f < 2; f++) {
      const _Float16* pa = lA + (wm * 32 + f * 16 + r) * LDA + ls * 8;
      ((uint4*)&af[f])[0] = *(const uint4*)(pa);
      ((uint4*)&af[f])[1] = *(const uint4*)(pa + 16);
    }
    #pragma unroll
    for (int s = 0; s < 2; s++) {
      v16h bf;
      const _Float16* pb = lB + (wn * 32 + s * 16 + r) * LDA + ls * 8;
      ((uint4*)&bf)[0] = *(const uint4*)(pb);
      ((uint4*)&bf)[1] = *(const uint4*)(pb + 16);
      #pragma unroll
      for (int f = 0; f < 2; f++)
        acc[f][s] = __builtin_amdgcn_wmma_f32_16x16x32_f16(
            false, af[f], false, bf, (short)0, acc[f][s], false, false);
    }
  }
  const int ls = lane >> 4, nl = lane & 15;
  #pragma unroll
  for (int f = 0; f < 2; f++)
    #pragma unroll
    for (int s = 0; s < 2; s++) {
      int o = nblk + wn * 32 + s * 16 + nl;
      float b = bias[o];
      #pragma unroll
      for (int j = 0; j < 8; j++) {
        int row = mblk + wm * 32 + f * 16 + ls * 8 + j;
        Y[(size_t)row * COUT + o] = acc[f][s][j] + b;
      }
    }
}

// ---------------------------------------------------------------------------
// BN stats: Y [Mrows, C] f32 -> stats[0..C)=sum, [C..2C)=sumsq (atomic accum)
// Requires gridDim.x*256 % C == 0 (C in {32,64,128}).
// ---------------------------------------------------------------------------
__global__ __launch_bounds__(256)
void bn_stats(const float* __restrict__ Y, float* __restrict__ stats,
              int Mrows, int C) {
  __shared__ float ssum[128], ssq[128];
  for (int i = threadIdx.x; i < C; i += 256) { ssum[i] = 0.f; ssq[i] = 0.f; }
  __syncthreads();
  size_t total  = (size_t)Mrows * C;
  size_t stride = (size_t)gridDim.x * 256;
  size_t g = (size_t)blockIdx.x * 256 + threadIdx.x;
  int c = (int)(g % (size_t)C);          // invariant across iterations
  float a = 0.f, b = 0.f;
  for (; g < total; g += stride) { float v = Y[g]; a += v; b += v * v; }
  atomicAdd(&ssum[c], a);
  atomicAdd(&ssq[c], b);
  __syncthreads();
  if (threadIdx.x < C) {
    atomicAdd(&stats[threadIdx.x],     ssum[threadIdx.x]);
    atomicAdd(&stats[C + threadIdx.x], ssq[threadIdx.x]);
  }
}

__global__ void bn_finalize(const float* __restrict__ stats,
                            const float* __restrict__ g,
                            const float* __restrict__ be,
                            float* __restrict__ invshift, int C, float invM) {
  int c = threadIdx.x;
  if (c >= C) return;
  float mean = stats[c] * invM;
  float var  = stats[C + c] * invM - mean * mean;
  float inv  = g[c] * rsqrtf(var + 1e-5f);
  invshift[c]     = inv;
  invshift[C + c] = be[c] - mean * inv;
}

// ---------------------------------------------------------------------------
// fused BN + LIF scan over T.  Y [T, NHWC] f32 -> spikes f16 [T, NHWC].
// LIF (tau=2, th=1, hard reset): v = 0.5*(v+z); spike = v>=1; v=0 on spike.
// ---------------------------------------------------------------------------
__global__ __launch_bounds__(256)
void bn_lif(const float* __restrict__ Y, const float* __restrict__ invshift,
            _Float16* __restrict__ Sout, int NHWC, int C) {
  int tid = blockIdx.x * 256 + threadIdx.x;
  if (tid >= NHWC) return;
  int c = tid % C;
  float inv = invshift[c], sh = invshift[C + c];
  float v = 0.f;
  #pragma unroll
  for (int t = 0; t < TSTEPS; t++) {
    float z = Y[(size_t)t * NHWC + tid] * inv + sh;
    v = 0.5f * (v + z);
    float spike = (v >= 1.0f) ? 1.0f : 0.0f;
    v = (v >= 1.0f) ? 0.0f : v;
    Sout[(size_t)t * NHWC + tid] = (_Float16)spike;
  }
}

// layer-1 variant: Y broadcast over T (single copy [NHWC])
__global__ __launch_bounds__(256)
void bn_lif_bcast(const float* __restrict__ Y, const float* __restrict__ invshift,
                  _Float16* __restrict__ Sout, int NHWC, int C) {
  int tid = blockIdx.x * 256 + threadIdx.x;
  if (tid >= NHWC) return;
  int c = tid % C;
  float z = Y[tid] * invshift[c] + invshift[C + c];
  float v = 0.f;
  #pragma unroll
  for (int t = 0; t < TSTEPS; t++) {
    v = 0.5f * (v + z);
    float spike = (v >= 1.0f) ? 1.0f : 0.0f;
    v = (v >= 1.0f) ? 0.0f : v;
    Sout[(size_t)t * NHWC + tid] = (_Float16)spike;
  }
}

// plain LIF (FC1 output): Y [T, NO] f32 -> spikes f16
__global__ __launch_bounds__(256)
void lif_plain(const float* __restrict__ Y, _Float16* __restrict__ Sout, int NO) {
  int tid = blockIdx.x * 256 + threadIdx.x;
  if (tid >= NO) return;
  float v = 0.f;
  #pragma unroll
  for (int t = 0; t < TSTEPS; t++) {
    float z = Y[(size_t)t * NO + tid];
    v = 0.5f * (v + z);
    float spike = (v >= 1.0f) ? 1.0f : 0.0f;
    v = (v >= 1.0f) ? 0.0f : v;
    Sout[(size_t)t * NO + tid] = (_Float16)spike;
  }
}

// ---------------------------------------------------------------------------
// 2x2 maxpool on f16 spikes (NHWC)
// ---------------------------------------------------------------------------
__global__ __launch_bounds__(256)
void maxpool2(const _Float16* __restrict__ Sin, _Float16* __restrict__ Sout,
              int TNimg, int H, int W, int C) {
  int Ho = H >> 1, Wo = W >> 1;
  size_t total = (size_t)TNimg * Ho * Wo * C;
  size_t tid = (size_t)blockIdx.x * 256 + threadIdx.x;
  if (tid >= total) return;
  int c = (int)(tid % (size_t)C);
  size_t r = tid / C;
  int wo = (int)(r % Wo); r /= Wo;
  int ho = (int)(r % Ho);
  int p  = (int)(r / Ho);
  const _Float16* base = Sin + (((size_t)p * H + ho * 2) * W + wo * 2) * C + c;
  float a = (float)base[0], b = (float)base[C];
  float d = (float)base[(size_t)W * C], e = (float)base[(size_t)W * C + C];
  Sout[tid] = (_Float16)fmaxf(fmaxf(a, b), fmaxf(d, e));
}

// ---------------------------------------------------------------------------
// FC2 (128->10) + LIF + mean over T.  Tiny -> scalar kernel.
// Hs [T*N,128] f16, W2 [10,128] f32, out [N,10] f32
// ---------------------------------------------------------------------------
__global__ __launch_bounds__(256)
void fc2_lif_mean(const _Float16* __restrict__ Hs, const float* __restrict__ W2,
                  const float* __restrict__ B2, float* __restrict__ Out) {
  int tid = blockIdx.x * 256 + threadIdx.x;
  if (tid >= NBATCH * 10) return;
  int o = tid % 10, n = tid / 10;
  float v = 0.f, acc = 0.f;
  for (int t = 0; t < TSTEPS; t++) {
    const _Float16* h = Hs + ((size_t)(t * NBATCH + n)) * 128;
    float z = B2[o];
    #pragma unroll 4
    for (int k = 0; k < 128; k++) z += (float)h[k] * W2[o * 128 + k];
    v = 0.5f * (v + z);
    if (v >= 1.0f) { acc += 1.0f; v = 0.0f; }
  }
  Out[tid] = acc * 0.125f;
}

// ===========================================================================
// host orchestration
// ===========================================================================
extern "C" void kernel_launch(void* const* d_in, const int* in_sizes, int n_in,
                              void* d_out, int out_size, void* d_ws, size_t ws_size,
                              hipStream_t stream) {
  const float* x    = (const float*)d_in[0];
  const float* w1   = (const float*)d_in[1];
  const float* g1   = (const float*)d_in[3];
  const float* be1  = (const float*)d_in[4];
  const float* wc[5]  = { (const float*)d_in[5],  (const float*)d_in[9],
                          (const float*)d_in[13], (const float*)d_in[17],
                          (const float*)d_in[21] };
  const float* gc[5]  = { (const float*)d_in[7],  (const float*)d_in[11],
                          (const float*)d_in[15], (const float*)d_in[19],
                          (const float*)d_in[23] };
  const float* bec[5] = { (const float*)d_in[8],  (const float*)d_in[12],
                          (const float*)d_in[16], (const float*)d_in[20],
                          (const float*)d_in[24] };
  const float* fc1_w = (const float*)d_in[25];
  const float* fc1_b = (const float*)d_in[26];
  const float* fc2_w = (const float*)d_in[27];
  const float* fc2_b = (const float*)d_in[28];
  float* out = (float*)d_out;

  // ---- workspace carve ----
  char* ws = (char*)d_ws;
  size_t off = 0;
  float*    Yb = (float*)(ws + off);    off += 134217728;   // 128 MiB pre-act
  _Float16* SA = (_Float16*)(ws + off); off += 67108864;    // s1 / s3 / s5
  _Float16* SB = (_Float16*)(ws + off); off += 67108864;    // s2 / s4 / s6
  _Float16* SP = (_Float16*)(ws + off); off += 16777216;    // pooled spikes
  float*    Hpre = (float*)(ws + off);  off += 524288;      // fc1 pre-LIF
  _Float16* Hs = (_Float16*)(ws + off); off += 262144;      // fc1 spikes
  _Float16* WT2 = (_Float16*)(ws + off);                    // packed weights
  _Float16* WT3 = WT2 + 32 * 288;
  _Float16* WT4 = WT3 + 64 * 288;
  _Float16* WT5 = WT4 + 64 * 576;
  _Float16* WT6 = WT5 + 128 * 576;
  _Float16* WTF = WT6 + 128 * 1152;     off += 2097152;
  float* stats    = (float*)(ws + off); off += 1024;        // sum|sumsq (<=2*128)
  float* invshift = (float*)(ws + off); off += 1024;        // inv|shift

  // ---- pack weights to f16 (im2col / NHWC-flatten order) ----
  pack_conv_w<<<(32 * 288 + 255) / 256, 256, 0, stream>>>(wc[0], WT2, 32, 32);
  pack_conv_w<<<(64 * 288 + 255) / 256, 256, 0, stream>>>(wc[1], WT3, 64, 32);
  pack_conv_w<<<(64 * 576 + 255) / 256, 256, 0, stream>>>(wc[2], WT4, 64, 64);
  pack_conv_w<<<(128 * 576 + 255) / 256, 256, 0, stream>>>(wc[3], WT5, 128, 64);
  pack_conv_w<<<(128 * 1152 + 255) / 256, 256, 0, stream>>>(wc[4], WT6, 128, 128);
  pack_fc1_w<<<(128 * 2048 + 255) / 256, 256, 0, stream>>>(fc1_w, WTF);

  // ---- layer 1: conv(3->32) once, BN, LIF scan over T ----
  conv1_direct<<<16384, 256, 0, stream>>>(x, w1, Yb);
  zero_f32<<<1, 256, 0, stream>>>(stats, 256);
  bn_stats<<<2048, 256, 0, stream>>>(Yb, stats, NBATCH * 1024, 32);
  bn_finalize<<<1, 128, 0, stream>>>(stats, g1, be1, invshift, 32, 1.f / (NBATCH * 1024));
  bn_lif_bcast<<<(NBATCH * 1024 * 32 + 255) / 256, 256, 0, stream>>>(
      Yb, invshift, SA, NBATCH * 1024 * 32, 32);

  // ---- layer 2: conv(32->32) @32x32, BN+LIF, pool -> SP [TN,16,16,32] ----
  conv3x3_wmma<32, 32><<<dim3(8192, 1), 256, 0, stream>>>(SA, WT2, Yb, TN * 1024, 32, 32);
  zero_f32<<<1, 256, 0, stream>>>(stats, 256);
  bn_stats<<<2048, 256, 0, stream>>>(Yb, stats, TN * 1024, 32);
  bn_finalize<<<1, 128, 0, stream>>>(stats, gc[0], bec[0], invshift, 32, 1.f / (TN * 1024.f));
  bn_lif<<<(NBATCH * 1024 * 32 + 255) / 256, 256, 0, stream>>>(
      Yb, invshift, SB, NBATCH * 1024 * 32, 32);
  maxpool2<<<(TN * 256 * 32 + 255) / 256, 256, 0, stream>>>(SB, SP, TN, 32, 32, 32);

  // ---- layer 3: conv(32->64) @16x16, BN+LIF -> SA ----
  conv3x3_wmma<32, 64><<<dim3(2048, 1), 256, 0, stream>>>(SP, WT3, Yb, TN * 256, 16, 16);
  zero_f32<<<1, 256, 0, stream>>>(stats, 256);
  bn_stats<<<2048, 256, 0, stream>>>(Yb, stats, TN * 256, 64);
  bn_finalize<<<1, 128, 0, stream>>>(stats, gc[1], bec[1], invshift, 64, 1.f / (TN * 256.f));
  bn_lif<<<(NBATCH * 256 * 64 + 255) / 256, 256, 0, stream>>>(
      Yb, invshift, SA, NBATCH * 256 * 64, 64);

  // ---- layer 4: conv(64->64) @16x16, BN+LIF, pool -> SP [TN,8,8,64] ----
  conv3x3_wmma<64, 64><<<dim3(2048, 1), 256, 0, stream>>>(SA, WT4, Yb, TN * 256, 16, 16);
  zero_f32<<<1, 256, 0, stream>>>(stats, 256);
  bn_stats<<<2048, 256, 0, stream>>>(Yb, stats, TN * 256, 64);
  bn_finalize<<<1, 128, 0, stream>>>(stats, gc[2], bec[2], invshift, 64, 1.f / (TN * 256.f));
  bn_lif<<<(NBATCH * 256 * 64 + 255) / 256, 256, 0, stream>>>(
      Yb, invshift, SB, NBATCH * 256 * 64, 64);
  maxpool2<<<(TN * 64 * 64 + 255) / 256, 256, 0, stream>>>(SB, SP, TN, 16, 16, 64);

  // ---- layer 5: conv(64->128) @8x8, BN+LIF -> SA ----
  conv3x3_wmma<64, 128><<<dim3(512, 2), 256, 0, stream>>>(SP, WT5, Yb, TN * 64, 8, 8);
  zero_f32<<<1, 256, 0, stream>>>(stats, 256);
  bn_stats<<<2048, 256, 0, stream>>>(Yb, stats, TN * 64, 128);
  bn_finalize<<<1, 128, 0, stream>>>(stats, gc[3], bec[3], invshift, 128, 1.f / (TN * 64.f));
  bn_lif<<<(NBATCH * 64 * 128 + 255) / 256, 256, 0, stream>>>(
      Yb, invshift, SA, NBATCH * 64 * 128, 128);

  // ---- layer 6: conv(128->128) @8x8, BN+LIF, pool -> SP [TN,4,4,128] ----
  conv3x3_wmma<128, 128><<<dim3(512, 2), 256, 0, stream>>>(SA, WT6, Yb, TN * 64, 8, 8);
  zero_f32<<<1, 256, 0, stream>>>(stats, 256);
  bn_stats<<<2048, 256, 0, stream>>>(Yb, stats, TN * 64, 128);
  bn_finalize<<<1, 128, 0, stream>>>(stats, gc[4], bec[4], invshift, 128, 1.f / (TN * 64.f));
  bn_lif<<<(NBATCH * 64 * 128 + 255) / 256, 256, 0, stream>>>(
      Yb, invshift, SB, NBATCH * 64 * 128, 128);
  maxpool2<<<(TN * 16 * 128 + 255) / 256, 256, 0, stream>>>(SB, SP, TN, 8, 8, 128);

  // ---- FC1 (2048->128) WMMA + bias, LIF ----
  fc1_wmma<<<dim3(8, 2), 256, 0, stream>>>(SP, WTF, fc1_b, Hpre);
  lif_plain<<<(NBATCH * 128 + 255) / 256, 256, 0, stream>>>(Hpre, Hs, NBATCH * 128);

  // ---- FC2 (128->10) + LIF + mean over T ----
  fc2_lif_mean<<<(NBATCH * 10 + 255) / 256, 256, 0, stream>>>(Hs, fc2_w, fc2_b, out);

  (void)in_sizes; (void)n_in; (void)out_size; (void)ws_size;
}
